// LineEDeeperGCN_1374389534971
// MI455X (gfx1250) — matrix operand order repaired and checked by reading, not compiled, not run on hardware
//
#include <hip/hip_runtime.h>
#include <hip/hip_bf16.h>
#include <math.h>

// ---------------- problem constants (from reference) ----------------
#define NN      10000
#define EE      80000
#define IN_F    64
#define HH      128
#define LL      4
#define OUT_F   10
#define CC      256          // 2*H
#define C2      512          // 2*C
#define ELINE   620000
#define EPS_M   1e-7f

// ---------------- CDNA5 WMMA types ----------------
typedef __attribute__((ext_vector_type(16))) __bf16 v16bf;
typedef __attribute__((ext_vector_type(8)))  float  v8f;

union FragBF { v16bf v; unsigned int u[8]; uint4 q[2]; };

// f32 -> bf16 (round to nearest even)
__device__ __forceinline__ unsigned short f2bf(float f) {
    unsigned int u = __float_as_uint(f);
    u += 0x7FFFu + ((u >> 16) & 1u);
    return (unsigned short)(u >> 16);
}

// float atomic max via ordered integer trick (logits here are always positive;
// general two-branch form kept for safety)
__device__ __forceinline__ void atomicMaxF(float* a, float v) {
    if (v >= 0.f) atomicMax((int*)a, __float_as_int(v));
    else          atomicMin((unsigned int*)a, (unsigned int)__float_as_int(v));
}

// ---------------- elementwise: f32 -> bf16 (row-major, for activations) ----------------
__global__ void k_cvt_bf16(const float* __restrict__ src,
                           unsigned short* __restrict__ dst, long n) {
    long i = (long)blockIdx.x * blockDim.x + threadIdx.x;
    long stride = (long)gridDim.x * blockDim.x;
    for (; i < n; i += stride) dst[i] = f2bf(src[i]);
}

// ---------------- weights: f32 [K,N] -> bf16 transposed [N,K] ----------------
// makes WMMA B-fragment loads fully contiguous (global_load_b128)
__global__ void k_cvt_bf16_T(const float* __restrict__ src,
                             unsigned short* __restrict__ dst, int K, int N) {
    long gid = (long)blockIdx.x * blockDim.x + threadIdx.x;
    long tot = (long)K * N;
    long stride = (long)gridDim.x * blockDim.x;
    for (; gid < tot; gid += stride) {
        int n = (int)(gid / K);
        int k = (int)(gid % K);
        dst[gid] = f2bf(src[(size_t)k * N + n]);   // coalesced write, strided read
    }
}

// ---------------- bf16 WMMA GEMM: C[M,N] = A[M,K] * BT^T + bias ----------------
// A: bf16 row-major [M,K]; BT: bf16 [N,K] (pre-transposed weights)
// one wave32 computes a 16x64 output strip: 4 accumulators share one A fragment
__global__ void k_gemm_wmma_bf16(const unsigned short* __restrict__ A,
                                 const unsigned short* __restrict__ BT,
                                 const float* __restrict__ bias,
                                 float* __restrict__ C,
                                 int M, int K, int N) {
    const int lane = threadIdx.x & 31;
    const int wid  = blockIdx.x * (blockDim.x >> 5) + (threadIdx.x >> 5);
    const int cg   = N >> 6;                 // 64-wide column groups
    const int total = (M >> 4) * cg;
    if (wid >= total) return;
    const int tm  = (wid / cg) << 4;
    const int tn0 = (wid % cg) << 6;

    // A fragment in this code's contiguous k order (B uses the same): lanes 0-15 -> K 0..15, lanes 16-31 -> K 16..31
    const int am  = tm + (lane & 15);
    const int akb = (lane & 16) ? 16 : 0;
    // B fragment in the same contiguous k order as A: lanes 0-15 -> K 0..15, lanes 16-31 -> K 16..31
    const int nl  = lane & 15;
    const int bkb = (lane & 16) ? 16 : 0;

    const unsigned short* ap  = A + (size_t)am * K + akb;
    const unsigned short* bp0 = BT + (size_t)(tn0 + nl)      * K + bkb;
    const unsigned short* bp1 = BT + (size_t)(tn0 + 16 + nl) * K + bkb;
    const unsigned short* bp2 = BT + (size_t)(tn0 + 32 + nl) * K + bkb;
    const unsigned short* bp3 = BT + (size_t)(tn0 + 48 + nl) * K + bkb;

    v8f acc0 = {}, acc1 = {}, acc2 = {}, acc3 = {};

    for (int k0 = 0; k0 < K; k0 += 32) {
        FragBF a;
        a.q[0] = *(const uint4*)(ap + k0);        // K pairs 0..7   (this lane's sub-base)
        a.q[1] = *(const uint4*)(ap + k0 + 8);    // K pairs 8..15
        __builtin_prefetch((const void*)(ap + k0 + 32), 0, 0); // global_prefetch_b8

        FragBF b0, b1, b2, b3;
        b0.q[0] = *(const uint4*)(bp0 + k0); b0.q[1] = *(const uint4*)(bp0 + k0 + 8);
        b1.q[0] = *(const uint4*)(bp1 + k0); b1.q[1] = *(const uint4*)(bp1 + k0 + 8);
        b2.q[0] = *(const uint4*)(bp2 + k0); b2.q[1] = *(const uint4*)(bp2 + k0 + 8);
        b3.q[0] = *(const uint4*)(bp3 + k0); b3.q[1] = *(const uint4*)(bp3 + k0 + 8);

        acc0 = __builtin_amdgcn_wmma_f32_16x16x32_bf16(false, a.v, false, b0.v, (short)0, acc0, false, false);
        acc1 = __builtin_amdgcn_wmma_f32_16x16x32_bf16(false, a.v, false, b1.v, (short)0, acc1, false, false);
        acc2 = __builtin_amdgcn_wmma_f32_16x16x32_bf16(false, a.v, false, b2.v, (short)0, acc2, false, false);
        acc3 = __builtin_amdgcn_wmma_f32_16x16x32_bf16(false, a.v, false, b3.v, (short)0, acc3, false, false);
    }

    // C/D 16x16 f32 layout: lanes 0-15 hold M=r, lanes 16-31 hold M=8+r
    const int mrb = tm + ((lane & 16) ? 8 : 0);
    v8f accs[4] = { acc0, acc1, acc2, acc3 };
    #pragma unroll
    for (int j = 0; j < 4; ++j) {
        const int cn = tn0 + j * 16 + nl;
        const float bv = bias[cn];
        #pragma unroll
        for (int r = 0; r < 8; ++r)
            C[(size_t)(mrb + r) * N + cn] = accs[j][r] + bv;
    }
}

// ---------------- z[e] = [h[src[e]] ; h[dst[e]]] ----------------
__global__ void k_concat(const float* __restrict__ h, const int* __restrict__ ei,
                         float* __restrict__ z) {
    int e = blockIdx.x;          // one block of 256 per edge
    int t = threadIdx.x;
    int s = ei[e];
    int d = ei[EE + e];
    float v = (t < HH) ? h[(size_t)s * HH + t] : h[(size_t)d * HH + (t - HH)];
    z[(size_t)e * CC + t] = v;
}

// ---------------- init per-layer softmax buffers ----------------
__global__ void k_seg_init(float* __restrict__ mx, float* __restrict__ den,
                           float* __restrict__ agg, long n) {
    long i = (long)blockIdx.x * blockDim.x + threadIdx.x;
    long stride = (long)gridDim.x * blockDim.x;
    for (; i < n; i += stride) {
        mx[i]  = -__builtin_huge_valf();
        den[i] = 0.f;
        agg[i] = 0.f;
    }
}

// ---------------- pass 1: per-dst segment max of logits ----------------
__global__ void k_seg_max(const float* __restrict__ zin, const int* __restrict__ lei,
                          const float* __restrict__ tptr, int layer,
                          float* __restrict__ mx) {
    int e = blockIdx.x;                 // one block of 64 per line-edge (float4 x 64 = 256)
    int t = threadIdx.x;
    int s = lei[e];
    int d = lei[ELINE + e];
    float tv = tptr[layer];
    float4 m = ((const float4*)(zin + (size_t)s * CC))[t];
    float lx = (fmaxf(m.x, 0.f) + EPS_M) * tv;
    float ly = (fmaxf(m.y, 0.f) + EPS_M) * tv;
    float lz = (fmaxf(m.z, 0.f) + EPS_M) * tv;
    float lw = (fmaxf(m.w, 0.f) + EPS_M) * tv;
    float* p = mx + (size_t)d * CC + t * 4;
    atomicMaxF(p + 0, lx);
    atomicMaxF(p + 1, ly);
    atomicMaxF(p + 2, lz);
    atomicMaxF(p + 3, lw);
}

// ---------------- pass 2: exp-sum and weighted message sum ----------------
__global__ void k_seg_sum(const float* __restrict__ zin, const int* __restrict__ lei,
                          const float* __restrict__ tptr, int layer,
                          const float* __restrict__ mx,
                          float* __restrict__ den, float* __restrict__ agg) {
    int e = blockIdx.x;
    int t = threadIdx.x;                // 0..63
    int s = lei[e];
    int d = lei[ELINE + e];
    float tv = tptr[layer];
    float4 m4 = ((const float4*)(zin + (size_t)s * CC))[t];
    float4 x4 = ((const float4*)(mx  + (size_t)d * CC))[t];
    float m[4] = { fmaxf(m4.x,0.f)+EPS_M, fmaxf(m4.y,0.f)+EPS_M,
                   fmaxf(m4.z,0.f)+EPS_M, fmaxf(m4.w,0.f)+EPS_M };
    float x[4] = { x4.x, x4.y, x4.z, x4.w };
    float* dp = den + (size_t)d * CC + t * 4;
    float* ap = agg + (size_t)d * CC + t * 4;
    #pragma unroll
    for (int j = 0; j < 4; ++j) {
        float ex = __expf(m[j] * tv - x[j]);
        atomicAdd(dp + j, ex);
        atomicAdd(ap + j, m[j] * ex);
    }
}

// ---------------- agg = agg/(den+1e-16) + x_r ----------------
__global__ void k_finalize(const float* __restrict__ zin, const float* __restrict__ den,
                           float* __restrict__ agg, long n) {
    long i = (long)blockIdx.x * blockDim.x + threadIdx.x;
    long stride = (long)gridDim.x * blockDim.x;
    for (; i < n; i += stride)
        agg[i] = agg[i] / (den[i] + 1e-16f) + zin[i];
}

// ---------------- LayerNorm + ReLU, one wave32 per row ----------------
__global__ void k_ln_relu(const float* __restrict__ in, const float* __restrict__ g,
                          const float* __restrict__ b, float* __restrict__ out,
                          int rows, int D) {
    int row  = blockIdx.x * 8 + (threadIdx.x >> 5);   // 8 waves per block
    int lane = threadIdx.x & 31;
    if (row >= rows) return;
    const float* p = in + (size_t)row * D;
    float s = 0.f;
    for (int c = lane; c < D; c += 32) s += p[c];
    #pragma unroll
    for (int o = 16; o; o >>= 1) s += __shfl_xor(s, o, 32);
    float mu = s / (float)D;
    float v = 0.f;
    for (int c = lane; c < D; c += 32) { float dd = p[c] - mu; v += dd * dd; }
    #pragma unroll
    for (int o = 16; o; o >>= 1) v += __shfl_xor(v, o, 32);
    float rs = rsqrtf(v / (float)D + 1e-5f);
    float* q = out + (size_t)row * D;
    for (int c = lane; c < D; c += 32) {
        float y = (p[c] - mu) * rs * g[c] + b[c];
        q[c] = y > 0.f ? y : 0.f;
    }
}

// ---------------- z += delta ----------------
__global__ void k_add(float* __restrict__ z, const float* __restrict__ d, long n) {
    long i = (long)blockIdx.x * blockDim.x + threadIdx.x;
    long stride = (long)gridDim.x * blockDim.x;
    for (; i < n; i += stride) z[i] += d[i];
}

// ---------------- out head: [E,256] @ [256,10] + b ----------------
__global__ void k_out_gemm(const float* __restrict__ hh, const float* __restrict__ w,
                           const float* __restrict__ b, float* __restrict__ out) {
    long gid = (long)blockIdx.x * blockDim.x + threadIdx.x;
    if (gid >= (long)EE * OUT_F) return;
    int e = (int)(gid / OUT_F);
    int o = (int)(gid % OUT_F);
    const float* row = hh + (size_t)e * CC;
    float acc = b[o];
    #pragma unroll 8
    for (int k = 0; k < CC; ++k) acc = fmaf(row[k], w[k * OUT_F + o], acc);
    out[gid] = acc;
}

// ================= host orchestration =================
static inline void gemm_launch(const unsigned short* A, const unsigned short* BT,
                               const float* bias, float* C,
                               int M, int K, int N, hipStream_t stream) {
    int waves = (M >> 4) * (N >> 6);               // 16x64 strip per wave
    int waves_per_block = 4;                       // 128 threads
    int grid = (waves + waves_per_block - 1) / waves_per_block;
    k_gemm_wmma_bf16<<<grid, waves_per_block * 32, 0, stream>>>(A, BT, bias, C, M, K, N);
}

extern "C" void kernel_launch(void* const* d_in, const int* in_sizes, int n_in,
                              void* d_out, int out_size, void* d_ws, size_t ws_size,
                              hipStream_t stream) {
    const float* x     = (const float*)d_in[0];
    const int*   ei    = (const int*)d_in[1];
    const int*   lei   = (const int*)d_in[3];
    const float* enc_w = (const float*)d_in[4];
    const float* enc_b = (const float*)d_in[5];
    const float* tarr  = (const float*)d_in[6];
    const float* W1    = (const float*)d_in[7];
    const float* b1    = (const float*)d_in[8];
    const float* ln1g  = (const float*)d_in[9];
    const float* ln1b  = (const float*)d_in[10];
    const float* W2    = (const float*)d_in[11];
    const float* b2    = (const float*)d_in[12];
    const float* ng    = (const float*)d_in[13];
    const float* nb    = (const float*)d_in[14];
    const float* ow    = (const float*)d_in[15];
    const float* ob    = (const float*)d_in[16];
    float* out = (float*)d_out;

    // scratch layout
    const size_t sz_h   = (size_t)NN * HH;
    const size_t sz_z   = (size_t)EE * CC;
    const size_t sz_mid = (size_t)EE * C2;
    float* h    = (float*)d_ws;
    float* z    = h   + sz_h;
    float* u    = z   + sz_z;      // conv input (hh) / final features
    float* mx   = u   + sz_z;
    float* den  = mx  + sz_z;
    float* agg  = den + sz_z;
    float* mid  = agg + sz_z;      // E x C2
    float* mout = mid + sz_mid;
    unsigned short* abf   = (unsigned short*)(mout + sz_z);   // activation bf16 (E x C2 max)
    unsigned short* w1bf  = abf  + sz_mid;                    // W1^T bf16 [C2, C]
    unsigned short* w2bf  = w1bf + (size_t)CC * C2;           // W2^T bf16 [C, C2]
    unsigned short* xbf   = w2bf + (size_t)CC * C2;           // x bf16 [N, IN_F]
    unsigned short* encbf = xbf  + (size_t)NN * IN_F;         // enc_w^T bf16 [H, IN_F]

    const int CB = 256;
    const int CG = 2048;

    // encoder: h = x @ enc_w + enc_b  (bf16 WMMA, transposed weights)
    k_cvt_bf16<<<CG, CB, 0, stream>>>(x, xbf, (long)NN * IN_F);
    k_cvt_bf16_T<<<64, CB, 0, stream>>>(enc_w, encbf, IN_F, HH);
    gemm_launch(xbf, encbf, enc_b, h, NN, IN_F, HH, stream);

    // line-graph nodes: z = [h[s]; h[d]]
    k_concat<<<EE, CC, 0, stream>>>(h, ei, z);

    for (int l = 0; l < LL; ++l) {
        const float* cin = z;
        if (l > 0) {
            // hh = relu(LN(z))
            k_ln_relu<<<(EE + 7) / 8, 256, 0, stream>>>(z, ng + (size_t)l * CC,
                                                        nb + (size_t)l * CC, u, EE, CC);
            cin = u;
        }
        // segment softmax aggregation over line edges
        k_seg_init<<<CG, CB, 0, stream>>>(mx, den, agg, (long)EE * CC);
        k_seg_max<<<ELINE, 64, 0, stream>>>(cin, lei, tarr, l, mx);
        k_seg_sum<<<ELINE, 64, 0, stream>>>(cin, lei, tarr, l, mx, den, agg);
        k_finalize<<<CG, CB, 0, stream>>>(cin, den, agg, (long)EE * CC);

        // MLP: Lin(C->C2) -> LN -> ReLU -> Lin(C2->C)
        k_cvt_bf16<<<CG, CB, 0, stream>>>(agg, abf, (long)EE * CC);
        k_cvt_bf16_T<<<512, CB, 0, stream>>>(W1 + (size_t)l * CC * C2, w1bf, CC, C2);
        gemm_launch(abf, w1bf, b1 + (size_t)l * C2, mid, EE, CC, C2, stream);
        k_ln_relu<<<(EE + 7) / 8, 256, 0, stream>>>(mid, ln1g + (size_t)l * C2,
                                                    ln1b + (size_t)l * C2, mid, EE, C2);
        k_cvt_bf16<<<CG, CB, 0, stream>>>(mid, abf, (long)EE * C2);
        k_cvt_bf16_T<<<512, CB, 0, stream>>>(W2 + (size_t)l * C2 * CC, w2bf, C2, CC);
        if (l == 0) {
            gemm_launch(abf, w2bf, b2 + (size_t)l * CC, z, EE, C2, CC, stream);
        } else {
            gemm_launch(abf, w2bf, b2 + (size_t)l * CC, mout, EE, C2, CC, stream);
            k_add<<<CG, CB, 0, stream>>>(z, mout, (long)EE * CC);
        }
    }

    // final: relu(LN(z, norm[0])) @ out_w + out_b
    k_ln_relu<<<(EE + 7) / 8, 256, 0, stream>>>(z, ng, nb, u, EE, CC);
    long tot = (long)EE * OUT_F;
    k_out_gemm<<<(int)((tot + CB - 1) / CB), CB, 0, stream>>>(u, ow, ob, out);
}
